// EquivariantMultiheadAttention_4140348473944
// MI455X (gfx1250) — compile-verified, matrix operand order checked by reading, not run on hardware
//
#include <hip/hip_runtime.h>
#include <hip/hip_bf16.h>
#include <math.h>

typedef __attribute__((ext_vector_type(16))) _Float16 v16h;
typedef __attribute__((ext_vector_type(8)))  float    v8f;

#define Bb 4
#define Tt 256
#define Ee 128
#define E3 384
#define Rr 64
#define WCOLS 512   // stage row: dk by channel [0,128) ; dv channel-major triplets [128,512)

#define AS3 __attribute__((address_space(3)))
static __device__ __forceinline__ unsigned lds_addr_of(void* p) {
    return (unsigned)(unsigned long long)(AS3 void*)p;
}

__device__ __forceinline__ float silu_f(float x) {
    // x * sigmoid(x) with fast hardware reciprocal (v_rcp_f32), not IEEE divide
    return x * __builtin_amdgcn_rcpf(1.0f + __expf(-x));
}

// ---------------------------------------------------------------------------
// Kernel 1: input projections. One block per token row (b*T + t), 128 threads.
// q (scaled), k, v (channel-major triplets), velocity_dot, w3 -> workspace.
// ---------------------------------------------------------------------------
__global__ __launch_bounds__(128)
void proj_kernel(const float* __restrict__ query, const float* __restrict__ velocity,
                 const float* __restrict__ Wq, const float* __restrict__ bq,
                 const float* __restrict__ Wk, const float* __restrict__ bk,
                 const float* __restrict__ Wv, const float* __restrict__ bv,
                 const float* __restrict__ Wvel, const float* __restrict__ bvel,
                 float* __restrict__ qbuf, float* __restrict__ kbuf,
                 float* __restrict__ vbufP, float* __restrict__ veldotbuf,
                 float* __restrict__ w3buf)
{
    __shared__ float qrow[Ee];
    __shared__ float vrow[3][Ee];
    const int row = blockIdx.x;           // b*T + t
    const int c   = threadIdx.x;          // 0..127 output channel
    qrow[c] = query[row * Ee + c];
    for (int x = 0; x < 3; ++x) vrow[x][c] = velocity[(row * 3 + x) * Ee + c];
    __syncthreads();

    // channel triplet columns of Wv for channel c = (h,d): h*48 + d + {0,16,32}
    const int h48 = (c >> 4) * 48;
    const int d   = c & 15;
    const int cv0 = h48 + d, cv1 = h48 + 16 + d, cv2 = h48 + 32 + d;

    float aq = 0.f, ak = 0.f, av0 = 0.f, av1 = 0.f, av2 = 0.f;
    for (int e = 0; e < Ee; ++e) {
        const float qe = qrow[e];
        aq  += qe * Wq[e * Ee + c];
        ak  += qe * Wk[e * Ee + c];
        av0 += qe * Wv[e * E3 + cv0];
        av1 += qe * Wv[e * E3 + cv1];
        av2 += qe * Wv[e * E3 + cv2];
    }
    qbuf[row * Ee + c] = (aq + bq[c]) * 0.25f;          // * d^-0.5, d=16
    kbuf[row * Ee + c] = ak + bk[c];
    float* vp = vbufP + ((size_t)row * Ee + c) * 3;     // channel-major triplet
    vp[0] = av0 + bv[cv0];
    vp[1] = av1 + bv[cv1];
    vp[2] = av2 + bv[cv2];

    float vdot = 0.f;
    for (int x = 0; x < 3; ++x) {
        float a1 = 0.f, a2 = 0.f, a3 = 0.f;
        for (int e = 0; e < Ee; ++e) {
            const float ve = vrow[x][e];
            a1 += ve * Wvel[e * E3 + c];
            a2 += ve * Wvel[e * E3 + 128 + c];
            a3 += ve * Wvel[e * E3 + 256 + c];
        }
        a1 += bvel[c]; a2 += bvel[128 + c]; a3 += bvel[256 + c];
        vdot += a1 * a2;
        w3buf[(row * 3 + x) * Ee + c] = a3;
    }
    veldotbuf[row * Ee + c] = vdot;
}

// ---------------------------------------------------------------------------
// Kernel 2: fused dk/dv WMMA GEMM + online-softmax attention + vector path.
// One block per (b, tq), 256 threads = 8 waves (4 col-tiles/wave -> Bfrag fits
// in 64 VGPRs, keeping total pressure under the 256-VGPR direct window).
// Double-buffered async global->LDS staging of edge_feature key tiles,
// v_wmma_f32_16x16x32_f16 GEMM, SiLU+bias into permuted LDS stage, then
// per-channel online softmax / masked reductions with 2-way key-split
// (threads c and c+128 each own half the keys; states merged once at end).
// ---------------------------------------------------------------------------
__global__ __launch_bounds__(256)
void attn_kernel(const float* __restrict__ edge_feature,
                 const float* __restrict__ edge_direction,
                 const float* __restrict__ cutoff,
                 const float* __restrict__ velocity,
                 const float* __restrict__ Wdk, const float* __restrict__ bdk,
                 const float* __restrict__ Wdv, const float* __restrict__ bdv,
                 const float* __restrict__ qbuf, const float* __restrict__ kbuf,
                 const float* __restrict__ vbufP,
                 float* __restrict__ attnbuf, float* __restrict__ vecbuf)
{
    __shared__ float Abuf[2][16 * 64];      // double-buffered key tile (f32, async dst)
    __shared__ float stage[16 * WCOLS];     // silu(ef @ [Wdk|Wdv] + b), dv permuted
    __shared__ float qrow[Ee];
    __shared__ float cutrow[Tt];
    __shared__ float edirL[3][Tt];

    const int row  = blockIdx.x;            // b*T + tq
    const int b    = row / Tt;
    const int tid  = threadIdx.x;           // 0..255
    const int lane = tid & 31;
    const int wv   = tid >> 5;              // wave 0..7
    const int m16  = lane & 15;             // A-row / B,D-col index within tile
    const int kh   = (lane >> 4) * 8;       // k-half select per ISA A/B layout

    const float* efrow = edge_feature + (size_t)row * Tt * Rr;

    // ---- prologue: async-copy key tile 0 into Abuf[0] (4KB, 16B per thread)
    {
        const unsigned l = lds_addr_of(&Abuf[0][0]) + tid * 16u;
        const unsigned long long g = (unsigned long long)(efrow + tid * 4);
        asm volatile("global_load_async_to_lds_b128 %0, %1, off"
                     :: "v"(l), "v"(g) : "memory");
    }

    if (tid < Ee) qrow[tid] = qbuf[row * Ee + tid];
    cutrow[tid] = cutoff[row * Tt + tid];
    for (int x = 0; x < 3; ++x)
        edirL[x][tid] = edge_direction[((size_t)row * Tt + tid) * 3 + x];

    // ---- Preload B fragments (weights) once: 4 column-tiles x 2 k-steps ----
    // Also precompute the permuted stage address + bias for this lane's D column.
    v16h  Bfrag[4][2];
    float biasv[4];
    int   saddr[4];
    #pragma unroll
    for (int i = 0; i < 4; ++i) {
        const int col = (wv * 4 + i) * 16 + m16;  // 0..511
        const float* Wp; int stride, cc;
        if (col < 128) {
            Wp = Wdk; stride = 128; cc = col; biasv[i] = bdk[cc];
            saddr[i] = col;                               // dk: by channel
        } else {
            Wp = Wdv; stride = 384; cc = col - 128; biasv[i] = bdv[cc];
            const int hh = cc / 48, jj = cc % 48;         // part = jj>>4, d = jj&15
            saddr[i] = 128 + (hh * 16 + (jj & 15)) * 3 + (jj >> 4);
        }
        #pragma unroll
        for (int ks = 0; ks < 2; ++ks) {
            const int kb = ks * 32;
            #pragma unroll
            for (int j = 0; j < 8; ++j) {
                Bfrag[i][ks][j]     = (_Float16)Wp[(kb + kh + j) * stride + cc];
                Bfrag[i][ks][8 + j] = (_Float16)Wp[(kb + 16 + kh + j) * stride + cc];
            }
        }
    }

    // ---- online-softmax + accumulation state: 2 threads per channel,
    //      thread `half` owns key rows [half*8, half*8+8) of every tile.
    const int c    = tid & 127;             // channel
    const int half = tid >> 7;              // 0 or 1
    float mrun = -3.402823466e+38f, srun = 0.f, accA = 0.f;
    float accV0 = 0.f, accV1 = 0.f, accV2 = 0.f;

    for (int tile = 0; tile < 16; ++tile) {
        const int cur = tile & 1;
        // ---- issue async copy of next tile into the other buffer, then wait
        if (tile + 1 < 16) {
            const unsigned l = lds_addr_of(&Abuf[cur ^ 1][0]) + tid * 16u;
            const unsigned long long g =
                (unsigned long long)(efrow + (size_t)(tile + 1) * 16 * Rr + tid * 4);
            asm volatile("global_load_async_to_lds_b128 %0, %1, off"
                         :: "v"(l), "v"(g) : "memory");
            asm volatile("s_wait_asynccnt 0x1" ::: "memory");  // current tile done
        } else {
            asm volatile("s_wait_asynccnt 0x0" ::: "memory");
        }
        __syncthreads();

        // ---- A fragments for this tile (f32 LDS -> f16 cvt at build time)
        const float* Ar = &Abuf[cur][m16 * 64];
        v16h af0, af1;
        #pragma unroll
        for (int j = 0; j < 8; ++j) {
            af0[j]     = (_Float16)Ar[kh + j];
            af0[8 + j] = (_Float16)Ar[16 + kh + j];
            af1[j]     = (_Float16)Ar[32 + kh + j];
            af1[8 + j] = (_Float16)Ar[48 + kh + j];
        }

        // ---- GEMM: each wave computes 4 [16x16] D tiles, silu+bias -> stage
        #pragma unroll
        for (int i = 0; i < 4; ++i) {
            v8f acc = {};
            acc = __builtin_amdgcn_wmma_f32_16x16x32_f16(
                      false, af0, false, Bfrag[i][0], (short)0, acc, false, false);
            acc = __builtin_amdgcn_wmma_f32_16x16x32_f16(
                      false, af1, false, Bfrag[i][1], (short)0, acc, false, false);
            #pragma unroll
            for (int r = 0; r < 8; ++r) {
                const int mrow = (lane >> 4) * 8 + r;   // D layout: VGPR r -> M
                stage[mrow * WCOLS + saddr[i]] = silu_f(acc[r] + biasv[i]);
            }
        }
        __syncthreads();

        // ---- elementwise phase: per-channel online softmax + reductions
        const int   tkbase = tile * 16 + half * 8;
        const float qv = qrow[c];
        for (int tl = 0; tl < 8; ++tl) {
            const int tk     = tkbase + tl;
            const int sl     = half * 8 + tl;           // row within stage
            const float cutv = cutrow[tk];
            const float dkv  = stage[sl * WCOLS + c];
            const float* dvp = &stage[sl * WCOLS + 128 + c * 3];   // contiguous triplet
            const int   krow = b * Tt + tk;
            const float kval = kbuf[krow * Ee + c];
            const float* vp  = vbufP + ((size_t)krow * Ee + c) * 3; // contiguous triplet
            const float val0 = vp[0] * dvp[0];
            const float val1 = vp[1] * dvp[1];
            const float val2 = vp[2] * dvp[2];

            const float aw = qv * kval * dkv * 0.25f;   // / sqrt(d)
            const float mn   = fmaxf(mrun, aw);
            const float corr = __expf(mrun - mn);
            const float p    = __expf(aw - mn);
            srun = srun * corr + p;
            accA = accA * corr + p * cutv * val0;
            mrun = mn;

            if (cutv != 0.f) {
                const float* vl = velocity + ((size_t)krow * 3) * Ee + c;
                accV0 += vl[0]      * val1 + edirL[0][tk] * val2;
                accV1 += vl[Ee]     * val1 + edirL[1][tk] * val2;
                accV2 += vl[2 * Ee] * val1 + edirL[2][tk] * val2;
            }
        }
        __syncthreads();   // protect stage before next tile's GEMM
    }

    // ---- merge the two per-channel softmax states (reuse stage as scratch)
    float* mb = stage;                      // 128 channels x 6 floats
    if (half == 1) {
        float* p = mb + c * 6;
        p[0] = mrun; p[1] = srun; p[2] = accA;
        p[3] = accV0; p[4] = accV1; p[5] = accV2;
    }
    __syncthreads();
    if (half == 0) {
        const float* p = mb + c * 6;
        const float m1 = p[0], s1 = p[1], A1 = p[2];
        const float mm = fmaxf(mrun, m1);
        const float c0 = __expf(mrun - mm), c1 = __expf(m1 - mm);
        const float s  = srun * c0 + s1 * c1;
        const float A  = accA * c0 + A1 * c1;
        attnbuf[row * Ee + c] =
            (s > 0.f) ? A * __builtin_amdgcn_rcpf(s) : 0.f;   // nan_to_num guard
        vecbuf[(row * 3 + 0) * Ee + c] = accV0 + p[3];
        vecbuf[(row * 3 + 1) * Ee + c] = accV1 + p[4];
        vecbuf[(row * 3 + 2) * Ee + c] = accV2 + p[5];
    }
}

// ---------------------------------------------------------------------------
// Kernel 3: output projection + assembly of dx / dvec.
// ---------------------------------------------------------------------------
__global__ __launch_bounds__(128)
void out_kernel(const float* __restrict__ Wo, const float* __restrict__ bo,
                const float* __restrict__ attnbuf, const float* __restrict__ veldotbuf,
                const float* __restrict__ w3buf, const float* __restrict__ vecbuf,
                float* __restrict__ out)
{
    __shared__ float arow[Ee];
    const int row = blockIdx.x;
    const int c   = threadIdx.x;
    arow[c] = attnbuf[row * Ee + c];
    __syncthreads();
    float o1 = 0.f, o2 = 0.f, o3 = 0.f;
    for (int e = 0; e < Ee; ++e) {
        const float a = arow[e];
        o1 += a * Wo[e * E3 + c];
        o2 += a * Wo[e * E3 + 128 + c];
        o3 += a * Wo[e * E3 + 256 + c];
    }
    o1 += bo[c]; o2 += bo[128 + c]; o3 += bo[256 + c];
    out[row * Ee + c] = veldotbuf[row * Ee + c] * o2 + o3;       // dx
    float* dvec = out + (size_t)Bb * Tt * Ee;
    for (int x = 0; x < 3; ++x)
        dvec[(row * 3 + x) * Ee + c] =
            w3buf[(row * 3 + x) * Ee + c] * o1 + vecbuf[(row * 3 + x) * Ee + c];
}

// ---------------------------------------------------------------------------
extern "C" void kernel_launch(void* const* d_in, const int* in_sizes, int n_in,
                              void* d_out, int out_size, void* d_ws, size_t ws_size,
                              hipStream_t stream) {
    const float* query   = (const float*)d_in[0];
    const float* velocity= (const float*)d_in[1];
    const float* edge_f  = (const float*)d_in[2];
    const float* edge_d  = (const float*)d_in[3];
    const float* cutoff  = (const float*)d_in[4];
    const float* Wq  = (const float*)d_in[5];  const float* bq  = (const float*)d_in[6];
    const float* Wk  = (const float*)d_in[7];  const float* bk  = (const float*)d_in[8];
    const float* Wv  = (const float*)d_in[9];  const float* bv  = (const float*)d_in[10];
    const float* Wo  = (const float*)d_in[11]; const float* bo  = (const float*)d_in[12];
    const float* Wvel= (const float*)d_in[13]; const float* bvel= (const float*)d_in[14];
    const float* Wdk = (const float*)d_in[15]; const float* bdk = (const float*)d_in[16];
    const float* Wdv = (const float*)d_in[17]; const float* bdv = (const float*)d_in[18];

    const int BT = Bb * Tt;                      // 1024
    float* ws = (float*)d_ws;
    float* qbuf    = ws;                 // BT*128
    float* kbuf    = qbuf    + BT * Ee;  // BT*128
    float* vbufP   = kbuf    + BT * Ee;  // BT*384 (channel-major triplets)
    float* veldot  = vbufP   + BT * E3;  // BT*128
    float* w3buf   = veldot  + BT * Ee;  // BT*384
    float* attnbuf = w3buf   + BT * E3;  // BT*128
    float* vecbuf  = attnbuf + BT * Ee;  // BT*384

    proj_kernel<<<BT, 128, 0, stream>>>(query, velocity, Wq, bq, Wk, bk, Wv, bv,
                                        Wvel, bvel, qbuf, kbuf, vbufP, veldot, w3buf);
    attn_kernel<<<BT, 256, 0, stream>>>(edge_f, edge_d, cutoff, velocity,
                                        Wdk, bdk, Wdv, bdv,
                                        qbuf, kbuf, vbufP, attnbuf, vecbuf);
    out_kernel<<<BT, 128, 0, stream>>>(Wo, bo, attnbuf, veldot, w3buf, vecbuf,
                                       (float*)d_out);
}
